// GATEncoder_82952998355437
// MI455X (gfx1250) — matrix operand order repaired and checked by reading, not compiled, not run on hardware
//
#include <hip/hip_runtime.h>
#include <cstdint>
#include <cstddef>

// ---------------------------------------------------------------------------
// GAT encoder for MI455X (gfx1250): bf16 WMMA for dense transforms (2x4 tile
// register blocking - no spills, b128 fragment loads, pre-packed B),
// CSR-gather segment softmax, fused BN+ELU epilogues.
// ---------------------------------------------------------------------------

typedef __attribute__((ext_vector_type(16))) __bf16 v16bf;
typedef __attribute__((ext_vector_type(8)))  __bf16 v8bf;
typedef __attribute__((ext_vector_type(8)))  float  v8f;

#define DEV __device__ __forceinline__

constexpr int   NN    = 32768;    // nodes
constexpr int   NPG   = 64;       // nodes per graph
constexpr int   BG    = 512;      // graphs
constexpr int   EE    = 524288;   // edges
constexpr int   INF_  = 13;
constexpr float NEG_SLOPE = 0.2f;
constexpr float BN_EPS    = 1e-5f;

DEV float lrelu(float x) { return x > 0.f ? x : NEG_SLOPE * x; }

// ---------------------------------------------------------------------------
// fp32 -> bf16 with optional column padding (dst_cols >= src_cols, pad = 0)
// ---------------------------------------------------------------------------
__global__ void cvt_padcols(const float* __restrict__ src, __bf16* __restrict__ dst,
                            int rows, int src_cols, int dst_cols) {
    int idx = blockIdx.x * blockDim.x + threadIdx.x;
    int total = rows * dst_cols;
    if (idx >= total) return;
    int r = idx / dst_cols, c = idx - r * dst_cols;
    float v = (c < src_cols) ? src[(size_t)r * src_cols + c] : 0.f;
    dst[idx] = (__bf16)v;
}

// ---------------------------------------------------------------------------
// Pack fp32 weights W[Ksrc, Ncol] into bf16 WMMA B-fragment order (K padded to
// Kpad, zero fill).  Layout: Bpack[((kt*tilesN + ct)*32 + lane)*16 + j] where
// b[j] <-> B[kt*32 + 16*(lane>>4) + j][ct*16 + (lane&15)].  Lets each lane in
// the GEMM fetch its whole fragment as two contiguous 16B loads.
// ---------------------------------------------------------------------------
__global__ void pack_w_bf16(const float* __restrict__ W, __bf16* __restrict__ Bpack,
                            int Ksrc, int Kpad, int Ncol) {
    int idx = blockIdx.x * blockDim.x + threadIdx.x;
    int tilesN = Ncol >> 4;
    int total = (Kpad >> 5) * tilesN * 512;
    if (idx >= total) return;
    int j    = idx & 15;
    int lane = (idx >> 4) & 31;
    int rest = idx >> 9;               // kt*tilesN + ct
    int ct   = rest % tilesN;
    int kt   = rest / tilesN;
    int k    = kt * 32 + 16 * (lane >> 4) + j;
    int col  = ct * 16 + (lane & 15);
    float v  = (k < Ksrc) ? W[(size_t)k * Ncol + col] : 0.f;
    Bpack[idx] = (__bf16)v;
}

// ---------------------------------------------------------------------------
// bf16 WMMA GEMM: C[M,Ncol] = A[M,K] * B[K,Ncol], fp32 accumulate.
// One wave computes a 32x64 block (2x4 tiles of 16x16): per K-step it issues
// 4 A + 8 B b128 loads, then 8 back-to-back v_wmma ops (A reused 4x, B 2x).
// 8 accs (64 VGPRs) + fragments stay well inside the 256-VGPR window so the
// compiler does not spill accumulators.  Tile id is wave-uniform -> EXEC
// stays all-ones for WMMA.  blockDim=(32,8).
// ---------------------------------------------------------------------------
union AFrag { v16bf v; v8bf h[2]; };

__global__ void __launch_bounds__(256, 1)
wmma_gemm_bf16(const __bf16* __restrict__ A,
               const __bf16* __restrict__ Bpack,
               float* __restrict__ C,
               int M, int Ncol, int K) {
    const int lane    = threadIdx.x;            // 0..31
    const int wave    = threadIdx.y;            // 0..7
    const int tilesN  = Ncol >> 4;
    const int cgroups = tilesN >> 2;            // groups of 4 column tiles
    const int rgroups = M >> 5;                 // groups of 2 row tiles
    const int tile    = blockIdx.x * 8 + wave;
    const int total   = rgroups * cgroups;
    if (tile >= total) return;                  // wave-uniform exit
    const int rg   = tile / cgroups;
    const int cg   = tile - rg * cgroups;
    const int rt0  = rg * 2;
    const int ct0  = cg * 4;
    const int row  = lane & 15;
    const int half = lane >> 4;

    const __bf16* Ab0 = A + (size_t)(rt0 * 16 + row) * K + 16 * half;
    v8f acc[8];
#pragma unroll
    for (int i = 0; i < 8; ++i) acc[i] = (v8f){};

    const int nkt = K >> 5;
    for (int kt = 0; kt < nkt; ++kt) {
        AFrag a[2];
#pragma unroll
        for (int r = 0; r < 2; ++r) {
            const __bf16* Ap = Ab0 + (size_t)(r * 16) * K + kt * 32;
            a[r].h[0] = *(const v8bf*)(Ap);        // K: 16*half .. +7
            a[r].h[1] = *(const v8bf*)(Ap + 8);    // K: 16*half+8 .. +15
        }
        const __bf16* Bb = Bpack + (size_t)((kt * tilesN + ct0) * 32 + lane) * 16;
        AFrag b[4];
#pragma unroll
        for (int q = 0; q < 4; ++q) {
            b[q].h[0] = *(const v8bf*)(Bb + q * 512);
            b[q].h[1] = *(const v8bf*)(Bb + q * 512 + 8);
        }
#pragma unroll
        for (int r = 0; r < 2; ++r)
#pragma unroll
            for (int q = 0; q < 4; ++q)
                acc[r * 4 + q] = __builtin_amdgcn_wmma_f32_16x16x32_bf16(
                    false, a[r].v, false, b[q].v, (short)0, acc[r * 4 + q],
                    false, false);
    }

    // C/D layout: lane col = lane&15, VGPR i -> M = 8*half + i
#pragma unroll
    for (int r = 0; r < 2; ++r) {
        float* Cb = C + (size_t)(rt0 * 16 + r * 16 + 8 * half) * Ncol + ct0 * 16 + row;
#pragma unroll
        for (int i = 0; i < 8; ++i) {
#pragma unroll
            for (int q = 0; q < 4; ++q)
                Cb[(size_t)i * Ncol + q * 16] = acc[r * 4 + q][i];
        }
    }
}

// ---------------------------------------------------------------------------
// Edge pass 1: in-degree counts + edge-attr sums per destination node
// ---------------------------------------------------------------------------
__global__ void edge_count(const int* __restrict__ dst, const float* __restrict__ ea,
                           int* __restrict__ deg_i, float* __restrict__ easum) {
    int e = blockIdx.x * blockDim.x + threadIdx.x;
    if (e >= EE) return;
    int d = dst[e];
    atomicAdd(&deg_i[d], 1);
    atomicAdd(&easum[d], ea[e]);
}

// Single-block exclusive scan of deg_i[0..NN) -> row_start[0..NN], 1024 thr
__global__ void scan_rowstart(const int* __restrict__ deg_i, int* __restrict__ row_start) {
    __shared__ int sdata[1024];
    __shared__ int carry_s;
    const int tid = threadIdx.x;
    if (tid == 0) carry_s = 0;
    __syncthreads();
    for (int base = 0; base < NN; base += 1024) {
        int v = deg_i[base + tid];
        sdata[tid] = v;
        __syncthreads();
        for (int off = 1; off < 1024; off <<= 1) {
            int x = (tid >= off) ? sdata[tid - off] : 0;
            __syncthreads();
            sdata[tid] += x;
            __syncthreads();
        }
        int incl = sdata[tid];
        int c = carry_s;
        __syncthreads();                        // everyone read carry
        row_start[base + tid] = c + incl - v;   // exclusive
        if (tid == 1023) carry_s = c + incl;
        __syncthreads();
    }
    if (tid == 0) row_start[NN] = carry_s;      // == EE
}

// Edge pass 2: bucket edge ids into CSR rows (order within a row irrelevant)
__global__ void csr_fill(const int* __restrict__ dst, const int* __restrict__ row_start,
                         int* __restrict__ cursor, int* __restrict__ csr_e) {
    int e = blockIdx.x * blockDim.x + threadIdx.x;
    if (e >= EE) return;
    int d = dst[e];
    int p = atomicAdd(&cursor[d], 1);
    csr_e[row_start[d] + p] = e;
}

// ---------------------------------------------------------------------------
// Per-head edge coefficient: k[h] = sum_c We[h*C+c] * att_edge[h*C+c]
// (edge_attr is 1-dim, so a_ed[e,h] = ea[e] * k[h])
// ---------------------------------------------------------------------------
__global__ void edge_coef(const float* __restrict__ We, const float* __restrict__ att_e,
                          float* __restrict__ kE, int H, int C) {
    int h = threadIdx.x;
    if (h >= H) return;
    float s = 0.f;
    for (int c = 0; c < C; ++c) s += We[h * C + c] * att_e[h * C + c];
    kE[h] = s;
}

// Per-(node,head): a_src, a_dst reductions + self-loop logit term
__global__ void attn_node(const float* __restrict__ hx,
                          const float* __restrict__ att_s, const float* __restrict__ att_d,
                          const float* __restrict__ easum, const int* __restrict__ deg_i,
                          const float* __restrict__ kE,
                          float* __restrict__ a_src, float* __restrict__ a_dst,
                          float* __restrict__ a_loop, int H, int C) {
    int idx = blockIdx.x * blockDim.x + threadIdx.x;
    if (idx >= NN * H) return;
    int n = idx / H, h = idx - n * H;
    const float* row = hx + (size_t)n * (H * C) + h * C;
    float s = 0.f, d = 0.f;
    for (int c = 0; c < C; ++c) {
        float v = row[c];
        s += v * att_s[h * C + c];
        d += v * att_d[h * C + c];
    }
    a_src[idx] = s;
    a_dst[idx] = d;
    float degf = fmaxf((float)deg_i[n], 1.f);
    a_loop[idx] = (easum[n] / degf) * kE[h];   // (ea_loop @ We) . att_edge
}

// ---------------------------------------------------------------------------
// Segment softmax + weighted aggregation per destination node (one WG/node).
// Fused: + bias, BatchNorm (inference), ELU, and bf16 write of the next
// layer's GEMM activations.  blockDim.x = H*C.
// ---------------------------------------------------------------------------
__global__ void aggregate(const float* __restrict__ hx,
                          const float* __restrict__ a_src, const float* __restrict__ a_dst,
                          const float* __restrict__ a_loop, const float* __restrict__ kE,
                          const int* __restrict__ srcArr, const float* __restrict__ ea,
                          const int* __restrict__ row_start, const int* __restrict__ csr_e,
                          const float* __restrict__ bias,
                          const float* __restrict__ bn_g, const float* __restrict__ bn_b,
                          const float* __restrict__ bn_m, const float* __restrict__ bn_v,
                          float* __restrict__ h_out, __bf16* __restrict__ h_out_bf,
                          int H, int C) {
    const int n  = blockIdx.x;
    const int t  = threadIdx.x;          // channel within H*C
    const int HC = H * C;
    const int h  = t / C;

    const float adst_n = a_dst[n * H + h];
    const float al = lrelu(a_src[n * H + h] + adst_n + a_loop[n * H + h]);
    const float kh = kE[h];
    const int rs = row_start[n], re = row_start[n + 1];

    // pass 1: segment max of logits (incl. self loop)
    float m = al;
    for (int i = rs; i < re; ++i) {
        int e = csr_e[i];
        int s = srcArr[e];
        float aev = lrelu(a_src[s * H + h] + adst_n + ea[e] * kh);
        m = fmaxf(m, aev);
        __builtin_prefetch(hx + (size_t)s * HC + t, 0, 0);   // global_prefetch_b8
    }

    // pass 2: exp-weighted gather
    float el  = __expf(al - m);
    float den = el;
    float acc = el * hx[(size_t)n * HC + t];
    for (int i = rs; i < re; ++i) {
        int e = csr_e[i];
        int s = srcArr[e];
        float aev = lrelu(a_src[s * H + h] + adst_n + ea[e] * kh);
        float w = __expf(aev - m);
        den += w;
        acc += w * hx[(size_t)s * HC + t];
    }

    float out = acc / den + bias[t];                       // concat=True (or H==1)
    // inference BatchNorm + ELU
    out = (out - bn_m[t]) * rsqrtf(bn_v[t] + BN_EPS) * bn_g[t] + bn_b[t];
    out = out > 0.f ? out : (__expf(out) - 1.f);
    h_out[(size_t)n * HC + t]    = out;
    h_out_bf[(size_t)n * HC + t] = (__bf16)out;            // next layer's A matrix
}

// ---------------------------------------------------------------------------
// Graph pooling: mean + max over the 64 nodes of each graph (contiguous ids)
// ---------------------------------------------------------------------------
__global__ void pool_meanmax(const float* __restrict__ h, float* __restrict__ pooled) {
    int b = blockIdx.x;            // graph
    int c = threadIdx.x;           // 0..127
    float s = 0.f, mx = -3.4e38f;
    const float* base = h + (size_t)b * NPG * 128 + c;
    for (int i = 0; i < NPG; ++i) {
        float v = base[(size_t)i * 128];
        s += v;
        mx = fmaxf(mx, v);
    }
    pooled[(size_t)b * 256 + c]       = s * (1.f / NPG);
    pooled[(size_t)b * 256 + 128 + c] = mx;
}

// MLP epilogue: + bias, BN without beta (beta terms cancel in reference), ReLU
__global__ void mlp_epilogue(const float* __restrict__ gm, const float* __restrict__ mb,
                             const float* __restrict__ mg, const float* __restrict__ mm,
                             const float* __restrict__ mv, float* __restrict__ out) {
    int idx = blockIdx.x * blockDim.x + threadIdx.x;
    if (idx >= BG * 128) return;
    int j = idx & 127;
    float v = gm[idx] + mb[j];
    v = (v - mm[j]) * rsqrtf(mv[j] + BN_EPS) * mg[j];
    out[idx] = fmaxf(v, 0.f);
}

// ---------------------------------------------------------------------------
// host side
// ---------------------------------------------------------------------------
static inline size_t alignup(size_t x) { return (x + 255) & ~(size_t)255; }

extern "C" void kernel_launch(void* const* d_in, const int* in_sizes, int n_in,
                              void* d_out, int out_size, void* d_ws, size_t ws_size,
                              hipStream_t stream) {
    (void)in_sizes; (void)n_in; (void)out_size; (void)ws_size;

    // ---- inputs (setup_inputs() insertion order, params flattened in order)
    const float* x_in = (const float*)d_in[0];
    const int*   ei   = (const int*)d_in[1];          // [0..E)=src, [E..2E)=dst
    const float* ea   = (const float*)d_in[2];
    // d_in[3] = batch (unused: node->graph mapping is contiguous)
    struct Conv { const float *W, *We, *as, *ad, *aeP, *b; };
    Conv c1{(const float*)d_in[4],  (const float*)d_in[5],  (const float*)d_in[6],
            (const float*)d_in[7],  (const float*)d_in[8],  (const float*)d_in[9]};
    Conv c2{(const float*)d_in[10], (const float*)d_in[11], (const float*)d_in[12],
            (const float*)d_in[13], (const float*)d_in[14], (const float*)d_in[15]};
    Conv c3{(const float*)d_in[16], (const float*)d_in[17], (const float*)d_in[18],
            (const float*)d_in[19], (const float*)d_in[20], (const float*)d_in[21]};
    const float* bn1[4] = {(const float*)d_in[22], (const float*)d_in[23],
                           (const float*)d_in[24], (const float*)d_in[25]};
    const float* bn2[4] = {(const float*)d_in[26], (const float*)d_in[27],
                           (const float*)d_in[28], (const float*)d_in[29]};
    const float* bn3[4] = {(const float*)d_in[30], (const float*)d_in[31],
                           (const float*)d_in[32], (const float*)d_in[33]};
    const float* mlpW = (const float*)d_in[34];
    const float* mlpB = (const float*)d_in[35];
    const float* mlpG = (const float*)d_in[36];
    const float* mlpM = (const float*)d_in[37];
    const float* mlpV = (const float*)d_in[38];

    const int* srcA = ei;
    const int* dstA = ei + EE;

    // ---- workspace carve (~90 MB)
    char* base = (char*)d_ws;
    size_t off = 0;
    auto carve = [&](size_t bytes) { char* p = base + off; off += alignup(bytes); return p; };
    __bf16* act      = (__bf16*)carve((size_t)NN * 256 * 2);   // bf16 activations
    __bf16* wb       = (__bf16*)carve(256 * 256 * 2);          // packed bf16 weights
    float*  hx       = (float*) carve((size_t)NN * 256 * 4);   // transformed features
    float*  hbuf     = (float*) carve((size_t)NN * 256 * 4);   // layer output (fp32)
    float*  a_src_b  = (float*) carve((size_t)NN * 4 * 4);
    float*  a_dst_b  = (float*) carve((size_t)NN * 4 * 4);
    float*  a_loop_b = (float*) carve((size_t)NN * 4 * 4);
    float*  kE       = (float*) carve(256);
    int*    deg_i    = (int*)   carve((size_t)NN * 4);         // | contiguous
    float*  easum    = (float*) carve((size_t)NN * 4);         // | zeroed
    int*    cursor   = (int*)   carve((size_t)NN * 4);         // | region
    int*    rowst    = (int*)   carve((size_t)(NN + 1) * 4);
    int*    csr_e    = (int*)   carve((size_t)EE * 4);
    float*  pooled   = (float*) carve((size_t)BG * 256 * 4);
    __bf16* pooled_b = (__bf16*)carve((size_t)BG * 256 * 2);
    float*  mlpo     = (float*) carve((size_t)BG * 128 * 4);

    // zero the atomic counters (deg_i | easum | cursor are contiguous, 256B-aligned)
    hipMemsetAsync(deg_i, 0, 3 * alignup((size_t)NN * 4), stream);

    const int TB = 256;
    auto g = [](int n, int b) { return (n + b - 1) / b; };

    // ---- CSR of incoming edges + per-node edge-attr mean precursors
    edge_count<<<g(EE, TB), TB, 0, stream>>>(dstA, ea, deg_i, easum);
    scan_rowstart<<<1, 1024, 0, stream>>>(deg_i, rowst);
    csr_fill<<<g(EE, TB), TB, 0, stream>>>(dstA, rowst, cursor, csr_e);

    dim3 gemmBlk(32, 8);
    auto gemm = [&](const __bf16* A, const __bf16* Bp, float* C, int M, int Ncol, int K) {
        int tiles = (M / 32) * (Ncol / 64);     // 2x4 tiles per wave
        wmma_gemm_bf16<<<g(tiles, 8), gemmBlk, 0, stream>>>(A, Bp, C, M, Ncol, K);
    };

    auto run_layer = [&](const Conv& cp, const float* const* bn,
                         int Kin, int Kpad, int H, int C) {
        int HC = H * C;
        pack_w_bf16<<<g(Kpad * HC, TB), TB, 0, stream>>>(cp.W, wb, Kin, Kpad, HC);
        gemm(act, wb, hx, NN, HC, Kpad);
        edge_coef<<<1, 32, 0, stream>>>(cp.We, cp.aeP, kE, H, C);
        attn_node<<<g(NN * H, TB), TB, 0, stream>>>(hx, cp.as, cp.ad, easum, deg_i, kE,
                                                    a_src_b, a_dst_b, a_loop_b, H, C);
        aggregate<<<NN, HC, 0, stream>>>(hx, a_src_b, a_dst_b, a_loop_b, kE,
                                         srcA, ea, rowst, csr_e, cp.b,
                                         bn[0], bn[1], bn[2], bn[3], hbuf, act, H, C);
    };

    // ---- three GAT layers (BN+ELU+bf16-recast fused into aggregation)
    cvt_padcols<<<g(NN * 32, TB), TB, 0, stream>>>(x_in, act, NN, INF_, 32);
    run_layer(c1, bn1, INF_, 32, 4, 64);   // 13 -> 4x64 (K padded 13->32)
    run_layer(c2, bn2, 256, 256, 4, 64);   // 256 -> 4x64  (act written by prev agg)
    run_layer(c3, bn3, 256, 256, 1, 128);  // 256 -> 1x128 (concat=False, H=1)

    // ---- pooling + MLP head (WMMA) + BN(no beta) + ReLU
    pool_meanmax<<<BG, 128, 0, stream>>>(hbuf, pooled);
    cvt_padcols<<<g(BG * 256, TB), TB, 0, stream>>>(pooled, pooled_b, BG, 256, 256);
    pack_w_bf16<<<g(256 * 128, TB), TB, 0, stream>>>(mlpW, wb, 256, 256, 128);
    gemm(pooled_b, wb, mlpo, BG, 128, 256);
    mlp_epilogue<<<g(BG * 128, TB), TB, 0, stream>>>(mlpo, mlpB, mlpG, mlpM, mlpV,
                                                     (float*)d_out);
}
